// AudioGaussianScene_58909771432694
// MI455X (gfx1250) — compile-verified, matrix operand order checked by reading, not run on hardware
//
#include <hip/hip_runtime.h>

// Problem dims (from reference)
#define N_GAUSS 512
#define T_DIM   1024
#define F_DIM   512

typedef __attribute__((ext_vector_type(16))) _Float16 v16h;
typedef __attribute__((ext_vector_type(8)))  _Float16 v8h;
typedef __attribute__((ext_vector_type(8)))  float    v8f;

// ---------------------------------------------------------------------------
// Phase 1: per-gaussian parameters.
//   at[g] = 0.5 / (sigma_t^2 * denom),  af[g] = 0.5 / (sigma_f^2 * denom)
// Also records which gaussians have rho != 0 (they need the exact VALU pass;
// for the reference inputs raw_rho == 0 so this list is empty).
// ---------------------------------------------------------------------------
__global__ __launch_bounds__(N_GAUSS)
void agsc_params_kernel(const float* __restrict__ log_sigma_t,
                        const float* __restrict__ log_sigma_f,
                        const float* __restrict__ raw_rho,
                        float* __restrict__ at, float* __restrict__ af,
                        int* __restrict__ corr_count, int* __restrict__ corr_idx) {
  const int g = threadIdx.x;
  if (g == 0) *corr_count = 0;
  __syncthreads();
  const float st    = __expf(log_sigma_t[g]);
  const float sf    = __expf(log_sigma_f[g]);
  const float rho   = tanhf(raw_rho[g]);
  const float denom = 1.0f - rho * rho + 1e-6f;
  at[g] = 0.5f / (st * st * denom);
  af[g] = 0.5f / (sf * sf * denom);
  if (rho != 0.0f) {
    const int slot = atomicAdd(corr_count, 1);
    corr_idx[slot] = g;
  }
}

// ---------------------------------------------------------------------------
// Phase 2a: Et[t][g] = exp(-at[g] * (t - mu_t[g])^2)   (f16, K-contiguous)
// ---------------------------------------------------------------------------
__global__ __launch_bounds__(256)
void agsc_factor_t_kernel(const float* __restrict__ t_grid,
                          const float* __restrict__ mu_t,
                          const float* __restrict__ at,
                          _Float16* __restrict__ Et) {
  const int idx = blockIdx.x * 256 + threadIdx.x;  // t * N_GAUSS + g
  const int t = idx >> 9;
  const int g = idx & (N_GAUSS - 1);
  const float d = t_grid[t] - mu_t[g];
  Et[idx] = (_Float16)__expf(-at[g] * d * d);
}

// ---------------------------------------------------------------------------
// Phase 2b: Ef[f][g] = alpha[g] * exp(-af[g] * (f - mu_f[g])^2)  (f16)
// ---------------------------------------------------------------------------
__global__ __launch_bounds__(256)
void agsc_factor_f_kernel(const float* __restrict__ f_grid,
                          const float* __restrict__ mu_f,
                          const float* __restrict__ af,
                          const float* __restrict__ raw_alpha,
                          _Float16* __restrict__ Ef) {
  const int idx = blockIdx.x * 256 + threadIdx.x;  // f * N_GAUSS + g
  const int f = idx >> 9;
  const int g = idx & (N_GAUSS - 1);
  const float d = f_grid[f] - mu_f[g];
  Ef[idx] = (_Float16)(raw_alpha[g] * __expf(-af[g] * d * d));
}

// ---------------------------------------------------------------------------
// Phase 3: out = Et (1024x512) x Ef^T (512x512) via v_wmma_f32_16x16x32_f16.
// One wave32 per 16x64 output strip: A fragment reused across 4 N-tiles,
// K=512 swept in 16 steps of 32. Fragment layouts per ISA §7.12.2:
//   A: lane(l<16)=row M=l, holds K {k0+0..7, k0+16..23}; lanes 16-31 hold
//      the same rows with K {k0+8..15, k0+24..31}. B symmetric over columns.
//   C/D: lane l<16 -> N=l, VGPR r -> M=r; lane l>=16 -> N=l-16, M=8+r.
// ---------------------------------------------------------------------------
__global__ __launch_bounds__(32)
void agsc_wmma_gemm_kernel(const _Float16* __restrict__ Et,
                           const _Float16* __restrict__ Ef,
                           float* __restrict__ out) {
  const int lane = threadIdx.x;
  const int row  = lane & 15;
  const int hi8  = (lane >> 4) * 8;       // 0 for lanes 0-15, 8 for 16-31
  const int m0   = blockIdx.x * 16;
  const int n0   = blockIdx.y * 64;

  v8f acc[4] = {v8f{}, v8f{}, v8f{}, v8f{}};

  const _Float16* arow = Et + (size_t)(m0 + row) * N_GAUSS;
  const _Float16* brow[4];
#pragma unroll
  for (int nt = 0; nt < 4; ++nt)
    brow[nt] = Ef + (size_t)(n0 + nt * 16 + row) * N_GAUSS;

  union Frag { v16h v; v8h h[2]; };

  for (int k0 = 0; k0 < N_GAUSS; k0 += 32) {
    Frag A;
    A.h[0] = *(const v8h*)(arow + k0 + hi8);
    A.h[1] = *(const v8h*)(arow + k0 + 16 + hi8);
#pragma unroll
    for (int nt = 0; nt < 4; ++nt) {
      Frag B;
      B.h[0] = *(const v8h*)(brow[nt] + k0 + hi8);
      B.h[1] = *(const v8h*)(brow[nt] + k0 + 16 + hi8);
      // 8 args: (neg_a, A, neg_b, B, c_mod, C, reuse_a, reuse_b)
      acc[nt] = __builtin_amdgcn_wmma_f32_16x16x32_f16(
          false, A.v, false, B.v, (short)0, acc[nt], false, false);
    }
  }

#pragma unroll
  for (int nt = 0; nt < 4; ++nt) {
    const int col = n0 + nt * 16 + row;
#pragma unroll
    for (int r = 0; r < 8; ++r)
      out[(size_t)(m0 + hi8 + r) * F_DIM + col] = acc[nt][r];
  }
}

// ---------------------------------------------------------------------------
// Phase 4: exact correction for gaussians with rho != 0 (replaces the
// separable approximation by the true bivariate term). For the reference
// inputs corr_count == 0, so every thread early-exits after one load.
// ---------------------------------------------------------------------------
__global__ __launch_bounds__(256)
void agsc_correction_kernel(const float* __restrict__ t_grid,
                            const float* __restrict__ f_grid,
                            const float* __restrict__ mu_t,
                            const float* __restrict__ mu_f,
                            const float* __restrict__ log_sigma_t,
                            const float* __restrict__ log_sigma_f,
                            const float* __restrict__ raw_rho,
                            const float* __restrict__ raw_alpha,
                            const int* __restrict__ corr_count,
                            const int* __restrict__ corr_idx,
                            float* __restrict__ out) {
  const int n = *corr_count;
  if (n == 0) return;
  const int idx = blockIdx.x * 256 + threadIdx.x;  // t * F_DIM + f
  const int t = idx >> 9;
  const int f = idx & (F_DIM - 1);
  const float tv = t_grid[t];
  const float fv = f_grid[f];
  float add = 0.0f;
  for (int i = 0; i < n; ++i) {
    const int g = corr_idx[i];
    const float st    = __expf(log_sigma_t[g]);
    const float sf    = __expf(log_sigma_f[g]);
    const float rho   = tanhf(raw_rho[g]);
    const float denom = 1.0f - rho * rho + 1e-6f;
    const float dt = (tv - mu_t[g]) / st;
    const float df = (fv - mu_f[g]) / sf;
    const float ztrue = (dt * dt - 2.0f * rho * dt * df + df * df) / denom;
    const float zsep  = (dt * dt + df * df) / denom;
    add += raw_alpha[g] * (__expf(-0.5f * ztrue) - __expf(-0.5f * zsep));
  }
  out[idx] += add;
}

// ---------------------------------------------------------------------------
// Launch
// ---------------------------------------------------------------------------
extern "C" void kernel_launch(void* const* d_in, const int* in_sizes, int n_in,
                              void* d_out, int out_size, void* d_ws, size_t ws_size,
                              hipStream_t stream) {
  (void)in_sizes; (void)n_in; (void)out_size; (void)ws_size;

  const float* t_grid      = (const float*)d_in[0];
  const float* f_grid      = (const float*)d_in[1];
  const float* mu_t        = (const float*)d_in[2];
  const float* mu_f        = (const float*)d_in[3];
  const float* log_sigma_t = (const float*)d_in[4];
  const float* log_sigma_f = (const float*)d_in[5];
  const float* raw_rho     = (const float*)d_in[6];
  const float* raw_alpha   = (const float*)d_in[7];
  float* out = (float*)d_out;

  // Workspace layout (~1.58 MB total)
  char* ws = (char*)d_ws;
  _Float16* Et = (_Float16*)ws;                                     // 1 MiB
  _Float16* Ef = (_Float16*)(ws + (size_t)T_DIM * N_GAUSS * 2);     // 512 KiB
  char* p = ws + (size_t)T_DIM * N_GAUSS * 2 + (size_t)F_DIM * N_GAUSS * 2;
  float* at       = (float*)p;  p += N_GAUSS * sizeof(float);
  float* af       = (float*)p;  p += N_GAUSS * sizeof(float);
  int*   corr_cnt = (int*)p;    p += sizeof(int);
  int*   corr_idx = (int*)p;

  agsc_params_kernel<<<1, N_GAUSS, 0, stream>>>(
      log_sigma_t, log_sigma_f, raw_rho, at, af, corr_cnt, corr_idx);

  agsc_factor_t_kernel<<<(T_DIM * N_GAUSS) / 256, 256, 0, stream>>>(
      t_grid, mu_t, at, Et);

  agsc_factor_f_kernel<<<(F_DIM * N_GAUSS) / 256, 256, 0, stream>>>(
      f_grid, mu_f, af, raw_alpha, Ef);

  dim3 gemm_grid(T_DIM / 16, F_DIM / 64);
  agsc_wmma_gemm_kernel<<<gemm_grid, 32, 0, stream>>>(Et, Ef, out);

  agsc_correction_kernel<<<(T_DIM * F_DIM) / 256, 256, 0, stream>>>(
      t_grid, f_grid, mu_t, mu_f, log_sigma_t, log_sigma_f,
      raw_rho, raw_alpha, corr_cnt, corr_idx, out);
}
